// CausalSelfAttention_13511967113429
// MI455X (gfx1250) — matrix operand order repaired and checked by reading, not compiled, not run on hardware
//
#include <hip/hip_runtime.h>

// Causal self-attention, fused, bf16 WMMA path for MI455X (gfx1250).
// B=2048, T=128, C=192, H=6, D=32.
//
// Workspace layout (bytes):
//   [0,         221184)     w_qkv  bf16, pre-swizzled to WMMA B-fragment order
//   [221184,    294912)     w_proj bf16, pre-swizzled
//   [294912,    +96MiB)     x      bf16  [B*T, C]
//   [next,      +288MiB)    qkv    bf16  [B,H,3,T,D]  (Q pre-scaled by 1/sqrt(D))
//   [next,      +96MiB)     y      bf16  [B*T, C]
// total ~480 MiB.

#define B_SZ 2048
#define T_SZ 128
#define C_SZ 192
#define H_SZ 6
#define D_SZ 32

typedef __attribute__((ext_vector_type(16))) __bf16        v16bf;
typedef __attribute__((ext_vector_type(8)))  float         v8f;
typedef __attribute__((ext_vector_type(4)))  float         f32x4;
typedef __attribute__((ext_vector_type(4)))  unsigned int  u32x4;
typedef __attribute__((ext_vector_type(2)))  unsigned int  u32x2;

__device__ __forceinline__ unsigned short bf16u(float f) {
  __bf16 b = (__bf16)f;                    // hardware f32->bf16 cvt
  return __builtin_bit_cast(unsigned short, b);
}

union Frag {
  v16bf v;
  unsigned short u[16];
  u32x4 q[2];
};

// ------------------------------------------------------------- x -> bf16 (vectorized)
__global__ void cvt_x4_kernel(const float* __restrict__ src,
                              unsigned short* __restrict__ dst, int n4) {
  int i = blockIdx.x * 256 + threadIdx.x;
  if (i >= n4) return;
  f32x4 f = ((const f32x4*)src)[i];
  union { unsigned short u[4]; u32x2 v; } o;
  o.u[0] = bf16u(f.x); o.u[1] = bf16u(f.y); o.u[2] = bf16u(f.z); o.u[3] = bf16u(f.w);
  ((u32x2*)dst)[i] = o.v;
}

// ------------------------------------------------------------- weight pre-swizzle
// dst[((ntile*6 + kstep)*32 + lane)*16 + i] = bf16( w[(kstep*32 + (lane>>4)*16 + i)*nout
//                                                    + ntile*16 + (lane&15)] )
// -> each lane's 16 B-fragment elements contiguous (two b128 loads).
__global__ void pack_w_kernel(const float* __restrict__ w,
                              unsigned short* __restrict__ dst, int nout) {
  int idx = blockIdx.x * 256 + threadIdx.x;
  int total = (nout / 16) * 6 * 32 * 16;
  if (idx >= total) return;
  int i     = idx & 15;
  int lane  = (idx >> 4) & 31;
  int kstep = (idx >> 9) % 6;
  int ntile = idx / (16 * 32 * 6);
  int col = ntile * 16 + (lane & 15);
  int k   = kstep * 32 + ((lane >> 4) << 4) + i;
  dst[idx] = bf16u(w[(size_t)k * nout + col]);
}

// ------------------------------------------------------------- GEMM (wave strip 64x16)
// A: [B*T, 192] bf16 row-major. W: pre-swizzled fragments. K = 192 = 6 WMMA steps.
// Each wave: 4 m-tiles stacked -> B fragment loaded once, used 4x (24 WMMAs/wave).
// QKV=true : scatter bf16 into [B,H,{q,k,v},T,D], Q scaled by 1/sqrt(D).
// QKV=false: f32 out + bias.
template <int NOUT, bool QKV>
__global__ void gemm_kernel(const unsigned short* __restrict__ A,
                            const unsigned short* __restrict__ wpk,
                            const float* __restrict__ bias,
                            void* __restrict__ outp) {
  const int NT = NOUT / 16;
  int wave = threadIdx.x >> 5;
  int lane = threadIdx.x & 31;
  int tile = blockIdx.x * 8 + wave;
  int m0 = (tile / NT) * 64;               // 64-row strip
  int n0 = (tile % NT) * 16;

  int koffA = (lane >> 4) << 4;            // 0 or 16
  int nc    = lane & 15;
  int col   = n0 + nc;
  const unsigned short* ar = A + (size_t)(m0 + nc) * C_SZ;
  const unsigned short* wt = wpk + ((size_t)(n0 >> 4) * 6 * 32 + lane) * 16;

  v8f acc[4] = {};
  #pragma unroll
  for (int ks = 0; ks < 6; ++ks) {
    Frag b;
    const unsigned short* bp = wt + ks * (32 * 16);
    b.q[0] = *(const u32x4*)(bp);
    b.q[1] = *(const u32x4*)(bp + 8);
    #pragma unroll
    for (int mi = 0; mi < 4; ++mi) {
      Frag a;
      const unsigned short* ap = ar + (size_t)mi * 16 * C_SZ + ks * 32 + koffA;
      a.q[0] = *(const u32x4*)(ap);
      a.q[1] = *(const u32x4*)(ap + 8);
      acc[mi] = __builtin_amdgcn_wmma_f32_16x16x32_bf16(false, a.v, false, b.v,
                                                        (short)0, acc[mi], false, false);
    }
  }

  float bv = bias[col];
  int thi  = (lane >> 4) << 3;             // 0 or 8 (C-layout row offset)
  if constexpr (QKV) {
    unsigned short* qkv = (unsigned short*)outp;
    int which = col / C_SZ;                // 0=q 1=k 2=v
    int rem   = col - which * C_SZ;
    int hh    = rem >> 5;
    int dd    = rem & 31;
    float postscale = (which == 0) ? 0.17677669529663687f : 1.0f; // fold 1/sqrt(D) into Q
    #pragma unroll
    for (int mi = 0; mi < 4; ++mi) {
      #pragma unroll
      for (int r = 0; r < 8; ++r) {
        int m  = m0 + mi * 16 + r + thi;
        int bb = m >> 7;                   // / T
        int tt = m & 127;
        size_t idx = ((((size_t)bb * H_SZ + hh) * 3 + which) * T_SZ + tt) * D_SZ + dd;
        qkv[idx] = bf16u((acc[mi][r] + bv) * postscale);
      }
    }
  } else {
    float* out = (float*)outp;
    #pragma unroll
    for (int mi = 0; mi < 4; ++mi) {
      #pragma unroll
      for (int r = 0; r < 8; ++r) {
        int m = m0 + mi * 16 + r + thi;
        out[(size_t)m * NOUT + col] = acc[mi][r] + bv;
      }
    }
  }
}

// ------------------------------------------------------------- fused attention
// One block per (b,h). 8 waves; wave w owns Q rows [16w,16w+16).
// Causal: wave w computes only S-column tiles j <= w (diagonal tile masked),
// P zero-padded to the 32-wide K-step boundary, PV runs ceil(16(w+1)/32) steps.
// LDS: sK[128*32] bf16 (8K) | sVt[32*128] bf16 (8K) | sS[128][128] f32 (64K).
// P (bf16) is written in place into the first half of each S row. Total 80K.
__global__ void attn_kernel(const unsigned short* __restrict__ qkv,
                            unsigned short* __restrict__ y) {
  extern __shared__ char smem[];
  unsigned short* sK  = (unsigned short*)smem;              // [128][32]
  unsigned short* sVt = sK + T_SZ * D_SZ;                   // [32][128]
  float*          sS  = (float*)(smem + 16384);             // [128][128]

  int tid  = threadIdx.x;
  int wave = tid >> 5, lane = tid & 31;
  int bh   = blockIdx.x;
  int bidx = bh / H_SZ, h = bh % H_SZ;
  size_t base = (size_t)bh * 3 * T_SZ * D_SZ;
  const unsigned short* Qg = qkv + base;
  const unsigned short* Kg = Qg + T_SZ * D_SZ;
  const unsigned short* Vg = Kg + T_SZ * D_SZ;

  for (int idx = tid; idx < T_SZ * D_SZ; idx += 256) {
    sK[idx] = Kg[idx];
    int s = idx >> 5, d = idx & 31;
    sVt[d * T_SZ + s] = Vg[idx];                            // V transposed
  }
  __syncthreads();

  int koffA = (lane >> 4) << 4;
  int koffB = (lane >> 4) << 4;
  int nc    = lane & 15;
  int thi   = (lane >> 4) << 3;

  // Q fragment (A, 16x32 bf16) straight from global (already scaled by 1/sqrt(D))
  int qrow = wave * 16 + nc;
  Frag qa;
  qa.q[0] = *(const u32x4*)(Qg + qrow * D_SZ + koffA);
  qa.q[1] = *(const u32x4*)(Qg + qrow * D_SZ + koffA + 8);

  // S = Q K^T, lower-triangle tiles only
  for (int j = 0; j <= wave; ++j) {
    Frag kb;                                                // B = K^T: contiguous row of K
    int srow = j * 16 + nc;
    kb.q[0] = *(const u32x4*)(sK + srow * D_SZ + koffB);
    kb.q[1] = *(const u32x4*)(sK + srow * D_SZ + koffB + 8);
    v8f acc = {};
    acc = __builtin_amdgcn_wmma_f32_16x16x32_bf16(false, qa.v, false, kb.v,
                                                  (short)0, acc, false, false);
    if (j == wave) {                                        // diagonal tile: mask
      #pragma unroll
      for (int r = 0; r < 8; ++r) {
        int t = wave * 16 + r + thi;
        int s = j * 16 + nc;
        sS[t * T_SZ + s] = (s > t) ? -3.0e38f : acc[r];
      }
    } else {
      #pragma unroll
      for (int r = 0; r < 8; ++r) {
        int t = wave * 16 + r + thi;
        sS[t * T_SZ + j * 16 + nc] = acc[r];
      }
    }
  }
  __syncthreads();

  // row softmax over valid prefix; P written bf16 in place, zero-padded to K-step
  int KT = (wave + 2) >> 1;                                 // PV K-steps for this wave
  if (lane < 16) {
    int t = wave * 16 + lane;
    int L = t + 1;                                          // valid columns
    int Lpad = KT * 32;
    float* rowp = sS + t * T_SZ;
    unsigned short* pp = (unsigned short*)rowp;             // first 256B of row
    float m = -3.0e38f;
    for (int s = 0; s < L; ++s) m = fmaxf(m, rowp[s]);
    float sum = 0.f;
    for (int s = 0; s < L; ++s) { float e = __expf(rowp[s] - m); rowp[s] = e; sum += e; }
    float inv = 1.0f / sum;
    for (int s = 0; s < L; ++s) pp[s] = bf16u(rowp[s] * inv);
    for (int s = L; s < Lpad; ++s) pp[s] = 0;
  }
  __syncthreads();

  // O = P @ V : KT WMMA K-steps, N=32 (2 tiles)
  v8f o0 = {}, o1 = {};
  const unsigned short* sProw = (const unsigned short*)(sS + (wave * 16 + nc) * T_SZ);
  for (int ki = 0; ki < KT; ++ki) {
    Frag pa;
    pa.q[0] = *(const u32x4*)(sProw + ki * 32 + koffA);
    pa.q[1] = *(const u32x4*)(sProw + ki * 32 + koffA + 8);
    Frag vb0, vb1;                                          // B = V: column d = row of sVt
    vb0.q[0] = *(const u32x4*)(sVt + nc * T_SZ + ki * 32 + koffB);
    vb0.q[1] = *(const u32x4*)(sVt + nc * T_SZ + ki * 32 + koffB + 8);
    vb1.q[0] = *(const u32x4*)(sVt + (nc + 16) * T_SZ + ki * 32 + koffB);
    vb1.q[1] = *(const u32x4*)(sVt + (nc + 16) * T_SZ + ki * 32 + koffB + 8);
    o0 = __builtin_amdgcn_wmma_f32_16x16x32_bf16(false, pa.v, false, vb0.v,
                                                 (short)0, o0, false, false);
    o1 = __builtin_amdgcn_wmma_f32_16x16x32_bf16(false, pa.v, false, vb1.v,
                                                 (short)0, o1, false, false);
  }

  #pragma unroll
  for (int r = 0; r < 8; ++r) {
    int t = wave * 16 + r + thi;
    size_t row = (size_t)bidx * T_SZ + t;
    unsigned short* yp = y + row * C_SZ + h * D_SZ;
    yp[nc]      = bf16u(o0[r]);
    yp[nc + 16] = bf16u(o1[r]);
  }
}

// ------------------------------------------------------------- launch
extern "C" void kernel_launch(void* const* d_in, const int* in_sizes, int n_in,
                              void* d_out, int out_size, void* d_ws, size_t ws_size,
                              hipStream_t stream) {
  const float* x      = (const float*)d_in[0];
  const float* w_qkv  = (const float*)d_in[1];
  const float* b_qkv  = (const float*)d_in[2];
  const float* w_proj = (const float*)d_in[3];
  const float* b_proj = (const float*)d_in[4];
  float* out = (float*)d_out;

  char* ws = (char*)d_ws;
  unsigned short* wqkv_pk  = (unsigned short*)ws;                          // 110592 elems
  unsigned short* wproj_pk = (unsigned short*)(ws + 221184);               // 36864 elems
  unsigned short* x_bf     = (unsigned short*)(ws + 294912);               // 50331648 elems
  unsigned short* qkv_bf   = (unsigned short*)(ws + 294912 + 100663296UL); // 150994944 elems
  unsigned short* y_bf     = (unsigned short*)(ws + 294912 + 100663296UL + 301989888UL);

  pack_w_kernel<<<(110592 + 255) / 256, 256, 0, stream>>>(w_qkv, wqkv_pk, 576);
  pack_w_kernel<<<(36864 + 255) / 256, 256, 0, stream>>>(w_proj, wproj_pk, 192);

  // x -> bf16, 4 elems/thread: 50331648/4 = 12582912 threads
  cvt_x4_kernel<<<12582912 / 256, 256, 0, stream>>>(x, x_bf, 12582912);

  // QKV: 4096 m-strips * 36 n-tiles = 147456 wave-strips / 8 waves per block
  gemm_kernel<576, true><<<147456 / 8, 256, 0, stream>>>(x_bf, wqkv_pk, b_qkv, qkv_bf);

  // fused attention: one block per (b,h), 80KB dynamic LDS
  attn_kernel<<<B_SZ * H_SZ, 256, 81920, stream>>>(qkv_bf, y_bf);

  // proj: 4096 m-strips * 12 n-tiles = 49152 wave-strips / 8
  gemm_kernel<192, false><<<49152 / 8, 256, 0, stream>>>(y_bf, wproj_pk, b_proj, out);
}
